// CausalAttention_65738769433303
// MI455X (gfx1250) — compile-verified, matrix operand order checked by reading, not run on hardware
//
#include <hip/hip_runtime.h>

typedef unsigned short u16;
typedef __attribute__((ext_vector_type(16))) __bf16 v16bf;
typedef __attribute__((ext_vector_type(8)))  float  v8f;
typedef __attribute__((ext_vector_type(4)))  unsigned int v4u;
typedef __attribute__((ext_vector_type(8)))  int v8i;
typedef __attribute__((ext_vector_type(4)))  int v4i;

#if __has_builtin(__builtin_amdgcn_tensor_load_to_lds)
#define HAVE_TDM 1
#else
#define HAVE_TDM 0
#endif

union Frag { v16bf v; uint4 q[2]; };

__device__ __forceinline__ u16 f2bf(float f) {
    union { float f; unsigned u; } x; x.f = f;
    unsigned u = x.u;
    u += 0x7FFFu + ((u >> 16) & 1u);   // round-to-nearest-even
    return (u16)(u >> 16);
}
__device__ __forceinline__ float bf2f(u16 h) {
    union { unsigned u; float f; } x; x.u = ((unsigned)h) << 16; return x.f;
}

__device__ __forceinline__ void wait_tensor0() {
#if __has_builtin(__builtin_amdgcn_s_wait_tensorcnt)
    __builtin_amdgcn_s_wait_tensorcnt(0);
#else
    asm volatile("s_wait_tensorcnt 0x0" ::: "memory");
#endif
}

#if HAVE_TDM
// DMA a rows x 64-half tile (row pitch strideElems halves) from global into LDS,
// stored contiguously row-major [rows][64]. Descriptor per CDNA5 D# (ISA 08_async_tensor).
__device__ __forceinline__ void tdm_load_tile(const u16* lds, const u16* g,
                                              unsigned strideElems, unsigned rows) {
    unsigned long long ga = (unsigned long long)(size_t)g;
    v4u g0;
    g0.x = 1u;                                                // count=1 (valid user desc)
    g0.y = (unsigned)(size_t)lds;                             // lds_addr
    g0.z = (unsigned)ga;                                      // global_addr[31:0]
    g0.w = (unsigned)((ga >> 32) & 0x01FFFFFFu) | (2u << 30); // global_addr[56:32] | type=2
    v8i g1;
    g1[0] = 0x00010000;            // workgroup_mask=0, data_size=1 (2 bytes)
    g1[1] = (int)(64u << 16);      // tensor_dim0 = 64   (bits 79:48)
    g1[2] = (int)(rows << 16);     // tensor_dim1 = rows (bits 111:80)
    g1[3] = (int)(64u << 16);      // tile_dim0 = 64     (bits 127:112)
    g1[4] = (int)rows;             // tile_dim1 = rows, tile_dim2 = 0
    g1[5] = (int)strideElems;      // tensor_dim0_stride[31:0]
    g1[6] = 0;
    g1[7] = 0;
    v4i z4 = {0, 0, 0, 0};
#if defined(__clang_major__) && __clang_major__ >= 23
    v8i z8 = {0, 0, 0, 0, 0, 0, 0, 0};
    __builtin_amdgcn_tensor_load_to_lds(g0, g1, z4, z4, z8, 0);
#else
    __builtin_amdgcn_tensor_load_to_lds(g0, g1, z4, z4, 0);
#endif
}
#endif

// Fragment from LDS buffer laid out [row][64 halves]; K-substep ks selects halves [ks*32, +32).
// Lanes 0-15: K {0..7,16..23}; lanes 16-31: K {8..15,24..31} of the substep.
__device__ __forceinline__ void ldsFrag64(Frag& f, const u16* buf, int row, int ks, int lane) {
    const u16* p = buf + row * 64 + ks * 32 + ((lane >> 4) << 3);
    f.q[0] = *(const uint4*)(p);
    f.q[1] = *(const uint4*)(p + 16);
}

#if !HAVE_TDM
// Fallback staging: all loads issued before stores.
__device__ __forceinline__ void stage_128x64(u16* dst, const u16* src, int srcLd, int t) {
    int row = t >> 1, seg = (t & 1) << 5;
    const uint4* g = (const uint4*)(src + (size_t)row * srcLd + seg);
    uint4 r0 = g[0], r1 = g[1], r2 = g[2], r3 = g[3];
    uint4* l = (uint4*)(dst + row * 64 + seg);
    l[0] = r0; l[1] = r1; l[2] = r2; l[3] = r3;
}
__device__ __forceinline__ void stage_256x64(u16* dst, const u16* src, int srcLd, int t) {
    const uint4* g = (const uint4*)(src + (size_t)t * srcLd);
    uint4 r[8];
#pragma unroll
    for (int i = 0; i < 8; ++i) r[i] = g[i];
    uint4* l = (uint4*)(dst + t * 64);
#pragma unroll
    for (int i = 0; i < 8; ++i) l[i] = r[i];
}
#endif

#define A_ROWS 128
#define B_ROWS 256
#define A_BUF (A_ROWS * 64)
#define B_BUF (B_ROWS * 64)

// NT GEMM core: acc[m,n] += sum_k A[m,k]*B[n,k]. Block tile 128(M)x256(N), K-step 64.
// 8 waves as 2(M)x4(N); each wave owns a 64x64 subtile: acc[4][4] 16x16 accumulators,
// so each 32-K substep is 16 WMMAs from 16 fragment loads (8 A + 8 B ds_load_b128).
__device__ __forceinline__ void gemm_core_nt(const u16* A, int lda, const u16* Bm, int ldb,
                                             int kTiles, u16* smA, u16* smB, v8f (&acc)[4][4]) {
    const int t = threadIdx.x, lane = t & 31, wv = t >> 5;
    const int wm = wv >> 2, wn = wv & 3;
#if HAVE_TDM
    if (wv == 0) {
        tdm_load_tile(smA, A, (unsigned)lda, A_ROWS);
        tdm_load_tile(smB, Bm, (unsigned)ldb, B_ROWS);
        wait_tensor0();
    }
    __syncthreads();
    for (int kt = 0; kt < kTiles; ++kt) {
        const int cur = kt & 1, nxt = cur ^ 1;
        if (wv == 0 && kt + 1 < kTiles) {   // prefetch next K-tile while computing this one
            tdm_load_tile(smA + nxt * A_BUF, A + (size_t)(kt + 1) * 64, (unsigned)lda, A_ROWS);
            tdm_load_tile(smB + nxt * B_BUF, Bm + (size_t)(kt + 1) * 64, (unsigned)ldb, B_ROWS);
        }
        const u16* bufA = smA + cur * A_BUF;
        const u16* bufB = smB + cur * B_BUF;
#pragma unroll
        for (int ks = 0; ks < 2; ++ks) {
            Frag a[4], b[4];
#pragma unroll
            for (int mt = 0; mt < 4; ++mt) ldsFrag64(a[mt], bufA, wm * 64 + mt * 16 + (lane & 15), ks, lane);
#pragma unroll
            for (int nt = 0; nt < 4; ++nt) ldsFrag64(b[nt], bufB, wn * 64 + nt * 16 + (lane & 15), ks, lane);
#pragma unroll
            for (int mt = 0; mt < 4; ++mt)
#pragma unroll
                for (int nt = 0; nt < 4; ++nt)
                    acc[mt][nt] = __builtin_amdgcn_wmma_f32_16x16x32_bf16(
                        false, a[mt].v, false, b[nt].v, (short)0, acc[mt][nt], false, false);
        }
        if (wv == 0 && kt + 1 < kTiles) wait_tensor0();
        __syncthreads();
    }
#else
    for (int kt = 0; kt < kTiles; ++kt) {
        stage_128x64(smA, A + (size_t)kt * 64, lda, t);
        stage_256x64(smB, Bm + (size_t)kt * 64, ldb, t);
        __syncthreads();
#pragma unroll
        for (int ks = 0; ks < 2; ++ks) {
            Frag a[4], b[4];
#pragma unroll
            for (int mt = 0; mt < 4; ++mt) ldsFrag64(a[mt], smA, wm * 64 + mt * 16 + (lane & 15), ks, lane);
#pragma unroll
            for (int nt = 0; nt < 4; ++nt) ldsFrag64(b[nt], smB, wn * 64 + nt * 16 + (lane & 15), ks, lane);
#pragma unroll
            for (int mt = 0; mt < 4; ++mt)
#pragma unroll
                for (int nt = 0; nt < 4; ++nt)
                    acc[mt][nt] = __builtin_amdgcn_wmma_f32_16x16x32_bf16(
                        false, a[mt].v, false, b[nt].v, (short)0, acc[mt][nt], false, false);
        }
        __syncthreads();
    }
#endif
}

// Visit each accumulator element with (row,col) inside the 128x256 block tile.
template <typename F>
__device__ __forceinline__ void epilogue(v8f (&acc)[4][4], F f) {
    int lane = threadIdx.x & 31, wv = threadIdx.x >> 5;
    int wm = wv >> 2, wn = wv & 3;
#pragma unroll
    for (int mt = 0; mt < 4; ++mt)
#pragma unroll
        for (int nt = 0; nt < 4; ++nt)
#pragma unroll
            for (int r = 0; r < 8; ++r) {
                int row = wm * 64 + mt * 16 + ((lane >> 4) << 3) + r;
                int col = wn * 64 + nt * 16 + (lane & 15);
                f(row, col, acc[mt][nt][r]);
            }
}

// ---------------- kernels ----------------

__global__ void __launch_bounds__(256) cvt4_f32_bf16(const float* __restrict__ src,
                                                     u16* __restrict__ dst, int n4) {
    int i = blockIdx.x * 256 + threadIdx.x;
    if (i >= n4) return;
    float4 v = ((const float4*)src)[i];
    union { uint2 q; u16 h[4]; } u;
    u.h[0] = f2bf(v.x); u.h[1] = f2bf(v.y); u.h[2] = f2bf(v.z); u.h[3] = f2bf(v.w);
    ((uint2*)dst)[i] = u.q;
}

// Y[m,n] = scale * sum_k A[m,k] * W[n,k]   (NT), bf16 out
__global__ void __launch_bounds__(256) qkv_gemm(const u16* __restrict__ X, const u16* __restrict__ W,
                                                u16* __restrict__ Y, int lda, int ldb, int ldc,
                                                int kTiles, float scale) {
    __shared__ u16 smA[2 * A_BUF];
    __shared__ u16 smB[2 * B_BUF];
    int m0 = blockIdx.y * 128, n0 = blockIdx.x * 256;
    v8f acc[4][4] = {};
    gemm_core_nt(X + (size_t)m0 * lda, lda, W + (size_t)n0 * ldb, ldb, kTiles, smA, smB, acc);
    epilogue(acc, [&](int row, int col, float v) {
        Y[(size_t)(m0 + row) * ldc + (n0 + col)] = f2bf(v * scale);
    });
}

// Sc[m,n] = sum_k Q[m,k]*K[n,k] with causal mask (n>m -> -inf), bf16 out. One batch per z.
__global__ void __launch_bounds__(256) scores_gemm(const u16* __restrict__ Q, const u16* __restrict__ K,
                                                   u16* __restrict__ Sc, int D, int S) {
    int b = blockIdx.z;
    const u16* A  = Q + (size_t)b * S * D;
    const u16* Bm = K + (size_t)b * S * D;
    u16* C = Sc + (size_t)b * S * S;
    int m0 = blockIdx.y * 128, n0 = blockIdx.x * 256;
    if (n0 > m0 + 127) {   // tile entirely above diagonal: all -inf, skip compute
        v8f z[4][4] = {};
        epilogue(z, [&](int row, int col, float) {
            C[(size_t)(m0 + row) * S + (n0 + col)] = 0xFF80u;  // bf16 -inf
        });
        return;
    }
    __shared__ u16 smA[2 * A_BUF];
    __shared__ u16 smB[2 * B_BUF];
    v8f acc[4][4] = {};
    gemm_core_nt(A + (size_t)m0 * D, D, Bm + (size_t)n0 * D, D, D / 64, smA, smB, acc);
    epilogue(acc, [&](int row, int col, float v) {
        int gr = m0 + row, gc = n0 + col;
        C[(size_t)gr * S + gc] = (gc > gr) ? (u16)0xFF80u : f2bf(v);
    });
}

// In-place row softmax over bf16 rows of length rowLen (=2048). One wave per row.
__global__ void __launch_bounds__(256) softmax_rows(u16* __restrict__ Sc, int rowLen) {
    int wave = threadIdx.x >> 5, lane = threadIdx.x & 31;
    size_t row = (size_t)blockIdx.x * 8 + wave;
    u16* p = Sc + row * rowLen;
    float vals[64];
    float m = -3.0e38f;
#pragma unroll
    for (int j = 0; j < 8; ++j) {
        union { uint4 q; u16 h[8]; } u;
        u.q = *(const uint4*)(p + j * 256 + lane * 8);
#pragma unroll
        for (int e = 0; e < 8; ++e) {
            float f = bf2f(u.h[e]);
            vals[j * 8 + e] = f;
            m = fmaxf(m, f);
        }
    }
#pragma unroll
    for (int off = 16; off >= 1; off >>= 1) m = fmaxf(m, __shfl_xor(m, off, 32));
    float s = 0.f;
#pragma unroll
    for (int i = 0; i < 64; ++i) { float e = __expf(vals[i] - m); vals[i] = e; s += e; }
#pragma unroll
    for (int off = 16; off >= 1; off >>= 1) s += __shfl_xor(s, off, 32);
    float inv = 1.0f / s;
#pragma unroll
    for (int j = 0; j < 8; ++j) {
        union { uint4 q; u16 h[8]; } u;
#pragma unroll
        for (int e = 0; e < 8; ++e) u.h[e] = f2bf(vals[j * 8 + e] * inv);
        *(uint4*)(p + j * 256 + lane * 8) = u.q;
    }
}

// 64x64 LDS tile transpose: src [rows][cols] -> dst [cols][rows], per batch z.
__global__ void __launch_bounds__(256) transpose_bf16(const u16* __restrict__ src, u16* __restrict__ dst,
                                                      int rows, int cols) {
    __shared__ u16 tile[64][65];
    int b = blockIdx.z;
    const u16* s = src + (size_t)b * rows * cols;
    u16* d = dst + (size_t)b * rows * cols;
    int r0 = blockIdx.y * 64, c0 = blockIdx.x * 64;
    int tr = threadIdx.x >> 2, seg = (threadIdx.x & 3) << 4;
    {
        union { uint4 q[2]; u16 h[16]; } u;
        const uint4* g = (const uint4*)(s + (size_t)(r0 + tr) * cols + c0 + seg);
        u.q[0] = g[0]; u.q[1] = g[1];
#pragma unroll
        for (int j = 0; j < 16; ++j) tile[tr][seg + j] = u.h[j];
    }
    __syncthreads();
    {
        union { uint4 q[2]; u16 h[16]; } u;
#pragma unroll
        for (int j = 0; j < 16; ++j) u.h[j] = tile[seg + j][tr];
        uint4* g = (uint4*)(d + (size_t)(c0 + tr) * rows + r0 + seg);
        g[0] = u.q[0]; g[1] = u.q[1];
    }
}

// O[m,n] = sum_k P[m,k]*Vt[n,k]  (NT, Vt pre-transposed [D][S]), f32 out.
// K-loop truncated at the causal boundary (P[m,k]==0 for k>m).
__global__ void __launch_bounds__(256) out_gemm(const u16* __restrict__ P, const u16* __restrict__ Vt,
                                                float* __restrict__ O, int S, int D) {
    int b = blockIdx.z;
    const u16* A  = P  + (size_t)b * S * S;
    const u16* Bm = Vt + (size_t)b * S * D;    // [D][S]
    float* C = O + (size_t)b * S * D;
    int m0 = blockIdx.y * 128, n0 = blockIdx.x * 256;
    int kTiles = m0 / 64 + 2;
    __shared__ u16 smA[2 * A_BUF];
    __shared__ u16 smB[2 * B_BUF];
    v8f acc[4][4] = {};
    gemm_core_nt(A + (size_t)m0 * S, S, Bm + (size_t)n0 * S, S, kTiles, smA, smB, acc);
    epilogue(acc, [&](int row, int col, float v) {
        C[(size_t)(m0 + row) * D + (n0 + col)] = v;
    });
}

extern "C" void kernel_launch(void* const* d_in, const int* in_sizes, int n_in,
                              void* d_out, int out_size, void* d_ws, size_t ws_size,
                              hipStream_t stream) {
    const int B = 4, S = 2048, D = 1024;
    const float* X  = (const float*)d_in[0];
    const float* Wq = (const float*)d_in[1];
    const float* Wk = (const float*)d_in[2];
    const float* Wv = (const float*)d_in[3];
    float* O = (float*)d_out;

    const size_t nX = (size_t)B * S * D;     // 8M
    const size_t nW = (size_t)D * D;         // 1M
    char* w = (char*)d_ws;
    u16* Xh  = (u16*)w; w += nX * 2;
    u16* Wqh = (u16*)w; w += nW * 2;
    u16* Wkh = (u16*)w; w += nW * 2;
    u16* Wvh = (u16*)w; w += nW * 2;
    u16* Qh  = (u16*)w; w += nX * 2;
    u16* Kh  = (u16*)w; w += nX * 2;
    u16* Vh  = (u16*)w; w += nX * 2;
    u16* Vt  = (u16*)w; w += nX * 2;
    u16* Sc  = (u16*)w;                      // B*S*S bf16 = 32 MB (L2-resident)

    dim3 blk(256);
    // 1) f32 -> bf16
    cvt4_f32_bf16<<<(int)((nX / 4 + 255) / 256), blk, 0, stream>>>(X, Xh, (int)(nX / 4));
    cvt4_f32_bf16<<<(int)((nW / 4 + 255) / 256), blk, 0, stream>>>(Wq, Wqh, (int)(nW / 4));
    cvt4_f32_bf16<<<(int)((nW / 4 + 255) / 256), blk, 0, stream>>>(Wk, Wkh, (int)(nW / 4));
    cvt4_f32_bf16<<<(int)((nW / 4 + 255) / 256), blk, 0, stream>>>(Wv, Wvh, (int)(nW / 4));

    // 2) QKV projections (scale 1/sqrt(D) folded into Q)
    dim3 gq(D / 256, (B * S) / 128);
    qkv_gemm<<<gq, blk, 0, stream>>>(Xh, Wqh, Qh, D, D, D, D / 64, 0.03125f);
    qkv_gemm<<<gq, blk, 0, stream>>>(Xh, Wkh, Kh, D, D, D, D / 64, 1.0f);
    qkv_gemm<<<gq, blk, 0, stream>>>(Xh, Wvh, Vh, D, D, D, D / 64, 1.0f);

    // 3) V transpose to [D][S] so P@V runs as NT through the same core
    transpose_bf16<<<dim3(D / 64, S / 64, B), blk, 0, stream>>>(Vh, Vt, S, D);

    // 4) causal-masked scores
    scores_gemm<<<dim3(S / 256, S / 128, B), blk, 0, stream>>>(Qh, Kh, Sc, D, S);

    // 5) row softmax (in place)
    softmax_rows<<<dim3(B * S / 8), blk, 0, stream>>>(Sc, S);

    // 6) O = P @ V
    out_gemm<<<dim3(D / 256, S / 128, B), blk, 0, stream>>>(Sc, Vt, O, S, D);
}
